// SlotModel_3204045603465
// MI455X (gfx1250) — matrix operand order repaired and checked by reading, not compiled, not run on hardware
//
#include <hip/hip_runtime.h>
#include <hip/hip_bf16.h>
#include <math.h>

typedef __attribute__((ext_vector_type(16))) _Float16 v16h;
typedef __attribute__((ext_vector_type(8)))  float    v8f;

#define HID   64
#define VOCAB 64
#define FF    128
#define BATCH 256
#define SEQL  4096
#define KEEP  4093   // L-3 positions eligible as slots
#define SLOTS 64

// ---------------------------------------------------------------------------
// Kernel A: per-vocab-token table. 1 block, 64 threads (thread v = token v).
// Computes hs_table (LayerNorm(e + FFN(e))), its row norms, then
//   G  = (hs@Wq + bq) @ hs^T          (64x64)   -> scores lookup
//   M2 = hs @ Wo                      (64x64)   -> folded output projection
//   order = vocab ids sorted by norm descending (stable)
// All exact f32; cost is negligible (64 rows).
// ---------------------------------------------------------------------------
__global__ __launch_bounds__(64)
void setup_tables(const float* __restrict__ embed,
                  const float* __restrict__ W1, const float* __restrict__ b1,
                  const float* __restrict__ W2, const float* __restrict__ b2,
                  const float* __restrict__ gamma, const float* __restrict__ beta,
                  const float* __restrict__ Wq, const float* __restrict__ bq,
                  const float* __restrict__ Wo,
                  float* __restrict__ G, float* __restrict__ M2,
                  int* __restrict__ order)
{
    __shared__ float sHS[VOCAB][HID + 1];   // +1 pad: kill bank conflicts
    __shared__ float sN[VOCAB];
    const int v = threadIdx.x;

    float e[HID];
#pragma unroll
    for (int i = 0; i < HID; ++i) e[i] = embed[v * HID + i];

    float f[HID];
#pragma unroll
    for (int i = 0; i < HID; ++i) f[i] = b2[i];

    for (int j = 0; j < FF; ++j) {
        float a = b1[j];
#pragma unroll
        for (int i = 0; i < HID; ++i) a += e[i] * W1[i * FF + j];
        a = fmaxf(a, 0.0f);
#pragma unroll
        for (int i = 0; i < HID; ++i) f[i] += a * W2[j * HID + i];
    }

    float mu = 0.0f;
#pragma unroll
    for (int i = 0; i < HID; ++i) { f[i] += e[i]; mu += f[i]; }
    mu *= (1.0f / HID);
    float var = 0.0f;
#pragma unroll
    for (int i = 0; i < HID; ++i) { float d = f[i] - mu; var += d * d; }
    var *= (1.0f / HID);
    const float rs = rsqrtf(var + 1e-5f);

    float nrm2 = 0.0f;
#pragma unroll
    for (int i = 0; i < HID; ++i) {
        float hs = (f[i] - mu) * rs * gamma[i] + beta[i];
        sHS[v][i] = hs;
        nrm2 += hs * hs;
    }
    sN[v] = sqrtf(nrm2);
    __syncthreads();

    // q-table row for token v
    float q[HID];
#pragma unroll
    for (int j = 0; j < HID; ++j) {
        float a = bq[j];
#pragma unroll
        for (int i = 0; i < HID; ++i) a += sHS[v][i] * Wq[i * HID + j];
        q[j] = a;
    }
    // G[v][u] = q_v . hs_u
    for (int u = 0; u < VOCAB; ++u) {
        float a = 0.0f;
#pragma unroll
        for (int i = 0; i < HID; ++i) a += q[i] * sHS[u][i];
        G[v * VOCAB + u] = a;
    }
    // M2[v][n] = hs_v . Wo[:,n]
    for (int n = 0; n < HID; ++n) {
        float a = 0.0f;
#pragma unroll
        for (int i = 0; i < HID; ++i) a += sHS[v][i] * Wo[i * HID + n];
        M2[v * HID + n] = a;
    }
    // stable descending rank by norm (distinct generically; ties -> index asc)
    {
        const float g = sN[v];
        int rank = 0;
        for (int u = 0; u < VOCAB; ++u)
            rank += (sN[u] > g) || (sN[u] == g && u < v);
        order[rank] = v;
    }
}

// ---------------------------------------------------------------------------
// Kernel B: per-batch token histogram (the only HBM-bound pass: 4 MB of seq),
// greedy slot fill in sorted-norm order, grouped softmax -> per-token weights.
// 256 blocks x 256 threads, coalesced loads.
// ---------------------------------------------------------------------------
__global__ __launch_bounds__(256)
void batch_weights(const int* __restrict__ seq,
                   const float* __restrict__ G,
                   const int* __restrict__ order,
                   float* __restrict__ Wmat)
{
    __shared__ int   cnt[VOCAB];
    __shared__ int   tv[VOCAB];
    __shared__ float red[VOCAB];
    __shared__ float smax, ssum;

    const int b   = blockIdx.x;
    const int tid = threadIdx.x;
    const int* s  = seq + (size_t)b * SEQL;

    if (tid < VOCAB) cnt[tid] = 0;
    __syncthreads();
    for (int p = tid; p < KEEP; p += 256)
        atomicAdd(&cnt[s[p]], 1);
    __syncthreads();

    if (tid == 0) {
        int rem = SLOTS;
        for (int k = 0; k < VOCAB; ++k) {
            const int v = order[k];
            const int t = (cnt[v] < rem) ? cnt[v] : rem;
            rem  -= t;
            tv[v] = t;
        }
    }
    __syncthreads();

    const int tlast = s[SEQL - 1];
    float sc = 0.0f;
    if (tid < VOCAB) {
        sc = G[tlast * VOCAB + tid] * 0.125f;           // 1/sqrt(64)
        red[tid] = (tv[tid] > 0) ? sc : -INFINITY;
    }
    __syncthreads();
    if (tid == 0) {
        float m = red[0];
        for (int u = 1; u < VOCAB; ++u) m = fmaxf(m, red[u]);
        smax = m;
    }
    __syncthreads();
    float ex = 0.0f;
    if (tid < VOCAB) {
        ex = (float)tv[tid] * expf(sc - smax);
        red[tid] = ex;
    }
    __syncthreads();
    if (tid == 0) {
        float t = 0.0f;
        for (int u = 0; u < VOCAB; ++u) t += red[u];
        ssum = t;
    }
    __syncthreads();
    if (tid < VOCAB)
        Wmat[(size_t)b * VOCAB + tid] = ex / ssum;
}

// ---------------------------------------------------------------------------
// Kernel C: out = Wmat(256x64) @ M2(64x64) + bo  via v_wmma_f32_16x16x32_f16.
// 16 blocks x 128 threads (4 waves); each wave owns one 16x16 output tile,
// K=64 -> two WMMA steps. EXEC all-ones, no divergence (WMMA requirement).
// Layouts per CDNA5 ISA §7.12.2.
// ---------------------------------------------------------------------------
__global__ __launch_bounds__(128)
void gemm_out(const float* __restrict__ Wmat,
              const float* __restrict__ M2,
              const float* __restrict__ bo,
              float* __restrict__ out)
{
    const int lane = threadIdx.x & 31;
    const int wave = threadIdx.x >> 5;
    const int grp  = lane >> 4;      // lane group 0/1
    const int mr   = lane & 15;      // row (A) / col (B,D) within tile
    const int m0   = blockIdx.x * 16;
    const int n0   = wave * 16;

    v8f c = {};
#pragma unroll
    for (int kb = 0; kb < HID; kb += 32) {
        v16h a, bm;
#pragma unroll
        for (int j = 0; j < 16; ++j) {
            // A element (mr, k): k = (j&7) + (j>=8 ? 16 : 0) + 8*grp
            const int ka = kb + (j & 7) + ((j >= 8) ? 16 : 0) + 8 * grp;
            a[j] = (_Float16)Wmat[(m0 + mr) * HID + ka];
            // B element (k, n0+mr): k = (j&7) + (j>=8 ? 16 : 0) + 8*grp
            const int kbv = kb + (j & 7) + ((j >= 8) ? 16 : 0) + 8 * grp;
            bm[j] = (_Float16)M2[kbv * HID + (n0 + mr)];
        }
        c = __builtin_amdgcn_wmma_f32_16x16x32_f16(
                /*neg_a=*/false, a, /*neg_b=*/false, bm,
                /*c_mod=*/(short)0, c, /*reuse_a=*/false, /*reuse_b=*/false);
    }
#pragma unroll
    for (int r = 0; r < 8; ++r) {
        const int m = m0 + r + 8 * grp;
        const int n = n0 + mr;
        out[m * HID + n] = c[r] + bo[n];
    }
}

// ---------------------------------------------------------------------------
extern "C" void kernel_launch(void* const* d_in, const int* in_sizes, int n_in,
                              void* d_out, int out_size, void* d_ws, size_t ws_size,
                              hipStream_t stream) {
    (void)in_sizes; (void)n_in; (void)out_size; (void)ws_size;
    const int*   seq   = (const int*)  d_in[0];
    const float* embed = (const float*)d_in[1];
    const float* W1    = (const float*)d_in[2];
    const float* b1    = (const float*)d_in[3];
    const float* W2    = (const float*)d_in[4];
    const float* b2    = (const float*)d_in[5];
    const float* gamma = (const float*)d_in[6];
    const float* beta  = (const float*)d_in[7];
    const float* Wq    = (const float*)d_in[8];
    const float* bq    = (const float*)d_in[9];
    const float* Wo    = (const float*)d_in[10];
    const float* bo    = (const float*)d_in[11];
    float* outp = (float*)d_out;

    // workspace layout (floats): G[4096] | M2[4096] | Wmat[16384] | order[64]
    float* G     = (float*)d_ws;
    float* M2    = G + VOCAB * VOCAB;
    float* Wmat  = M2 + VOCAB * HID;
    int*   order = (int*)(Wmat + BATCH * VOCAB);

    setup_tables<<<1, 64, 0, stream>>>(embed, W1, b1, W2, b2, gamma, beta,
                                       Wq, bq, Wo, G, M2, order);
    batch_weights<<<BATCH, 256, 0, stream>>>(seq, G, order, Wmat);
    gemm_out<<<BATCH / 16, 128, 0, stream>>>(Wmat, M2, bo, outp);
}
